// MotionEncoder_22986664968870
// MI455X (gfx1250) — matrix operand order repaired and checked by reading, not compiled, not run on hardware
//
#include <hip/hip_runtime.h>

// ---------------------------------------------------------------------------
// MotionEncoder for MI455X (gfx1250, wave32, WMMA + TDM).
//  (1) zero a halo-padded f16 occupancy grid (258x258x18x7 per batch, d_ws)
//  (2) scatter points: plain same-value u16 stores of half(0.5) - deterministic
//  (3) implicit-GEMM stencil conv: TENSOR_LOAD_TO_LDS pulls the haloed
//      3x18x18x7 tile into LDS in one DMA (s_wait_tensorcnt), then
//      3x v_wmma_f32_16x16x32_f16 per 16-voxel tile (M=16 vox, K=81->96,
//      N=8->16); head goes through a conflict-free LDS transpose so each
//      lane finishes one voxel with a coalesced store.
// ---------------------------------------------------------------------------

typedef __attribute__((ext_vector_type(16))) _Float16 v16h;
typedef __attribute__((ext_vector_type(8)))  float    v8f;
typedef __attribute__((ext_vector_type(4)))  unsigned int tdm_v4u;
typedef __attribute__((ext_vector_type(8)))  int          tdm_v8i;
typedef __attribute__((ext_vector_type(4)))  int          tdm_v4i;

#define BX   2
#define NPTS 300000
#define DX   256
#define DY   256
#define DZ   16
#define DT   5
#define VOX  (DX*DY*DZ*DT)
#define CMID 8

// halo-padded occupancy grid in d_ws (f16 values 0 / 0.5)
#define PADX 258
#define PADY 258
#define PADZ 18
#define PADT 7
#define PADVOX (PADX*PADY*PADZ*PADT)      // 8,387,064 halfs per batch
#define GRID_HALFS (BX*PADVOX)            // 16,774,128 halfs (~32 MB)
#define GRID_U128  (GRID_HALFS/8)         // exact: 2,096,766 uint4 stores

// LDS tile: x {x-1..x+1}, y strip 16+2, z 16+2, t 5+2  (t innermost)
#define TLY 18
#define TILE_ELEMS (3*18*18*7)            // 6804 halfs = 13,608 B
#define YSTRIDE (PADZ*PADT)               // 126
#define XSTRIDE (PADY*PADZ*PADT)          // 32508
#define SCR_STRIDE 20                     // floats; 80 B rows: 16B-aligned,
                                          // n*20 % 64 distinct -> no conflicts

__global__ __launch_bounds__(256)
void me_zero_grid(uint4* __restrict__ g) {
  int i = blockIdx.x * 256 + threadIdx.x;
  if (i < GRID_U128) g[i] = make_uint4(0u, 0u, 0u, 0u);
}

__global__ __launch_bounds__(256)
void me_scatter(const float* __restrict__ pts, unsigned short* __restrict__ gpad, int ntot) {
  int i = blockIdx.x * 256 + threadIdx.x;
  if (i >= ntot) return;
  int b = i / NPTS;
  const float* p = pts + (size_t)i * 4;
  int cx = (int)floorf(p[0] / 0.4f); cx = cx < 0 ? 0 : (cx > DX-1 ? DX-1 : cx);
  int cy = (int)floorf(p[1] / 0.4f); cy = cy < 0 ? 0 : (cy > DY-1 ? DY-1 : cy);
  int cz = (int)floorf(p[2] / 0.4f); cz = cz < 0 ? 0 : (cz > DZ-1 ? DZ-1 : cz);
  int ct = (int)floorf(p[3] / 1.0f); ct = ct < 0 ? 0 : (ct > DT-1 ? DT-1 : ct);
  size_t lin = (size_t)b*PADVOX +
               (size_t)(((cx+1)*PADY + (cy+1))*PADZ + (cz+1))*PADT + (ct+1);
  gpad[lin] = 0x3800;   // half(0.5); same-value race -> deterministic
}

__global__ __launch_bounds__(256)
void me_conv_mlp(const unsigned short* __restrict__ gpad,
                 const float* __restrict__ W0,   // (81,1,8)
                 const float* __restrict__ b0,   // (8)
                 const float* __restrict__ W1,   // (8,1)
                 const float* __restrict__ b1,   // (1)
                 float* __restrict__ out) {      // (B,1,X,Y,Z,T)
  __shared__ _Float16 tile[TILE_ELEMS];
  __shared__ float    scr[8 * 16 * SCR_STRIDE];  // per-wave C transpose buffer

  int blk = blockIdx.x;                 // b*(X*16) + x*16 + yb
  int yb  = blk & 15;
  int x   = (blk >> 4) & 255;
  int b   = blk >> 12;
  int y0  = yb * 16;

  __builtin_prefetch(W0, 0, 1);         // global_prefetch_b8 for the weights

  // tile origin in padded grid: (x, y0, 0, 0) in padded coords
  size_t lin0 = (size_t)b*PADVOX + (size_t)(x*PADY + y0)*YSTRIDE;

#if __has_builtin(__builtin_amdgcn_tensor_load_to_lds)
  // ---- TDM: one 4D tensor DMA (tile 7x18x18x3, f16) global -> LDS ----
  if (threadIdx.x < 32) {
    unsigned long long ga = (unsigned long long)(uintptr_t)gpad + 2ull*lin0;
    unsigned int ldsa = (unsigned int)(uintptr_t)(&tile[0]);   // LDS byte offset
    tdm_v4u g0; tdm_v8i g1; tdm_v4i g2, g3; tdm_v8i g4;
    // D# group 0: count=1 | lds_addr | global_addr(57b) | type=2
    g0[0] = 1u;
    g0[1] = ldsa;
    g0[2] = (unsigned int)(ga & 0xffffffffull);
    g0[3] = (unsigned int)((ga >> 32) & 0x01ffffffull) | (2u << 30);
    // D# group 1: wg_mask=0, data_size=1(2B), dims/strides in elements
    g1[0] = (int)(1u << 16);                       // data_size=2B
    g1[1] = (int)((unsigned)PADT << 16);           // tensor_dim0=7 (lo16)
    g1[2] = (int)((unsigned)PADZ << 16);           // dim0 hi=0 | tensor_dim1=18
    g1[3] = (int)((unsigned)PADT << 16);           // dim1 hi=0 | tile_dim0=7
    g1[4] = (int)(18u | (18u << 16));              // tile_dim1=18 | tile_dim2=18
    g1[5] = (int)PADT;                             // tensor_dim0_stride=7
    g1[6] = (int)((unsigned)YSTRIDE << 16);        // stride0 hi=0 | stride1=126
    g1[7] = 0;                                     // stride1 hi bits
    // D# group 2: tensor_dim2=258, tensor_dim3=258, stride2=32508, tile_dim3=3
    g2[0] = PADY;
    g2[1] = PADX;
    g2[2] = XSTRIDE;
    g2[3] = (int)(3u << 16);
    // D# group 3: unused (no dim4)
    g3[0] = 0; g3[1] = 0; g3[2] = 0; g3[3] = 0;
    // extra operand on the 6-arg toolchain: zeroed
    g4[0] = 0; g4[1] = 0; g4[2] = 0; g4[3] = 0;
    g4[4] = 0; g4[5] = 0; g4[6] = 0; g4[7] = 0;
    __builtin_amdgcn_tensor_load_to_lds(g0, g1, g2, g3, g4, 0);
    __builtin_amdgcn_s_wait_tensorcnt(0);
  }
  __syncthreads();
#else
  // ---- fallback: cooperative strided copy (grid is pre-padded, no checks) --
  for (int i = threadIdx.x; i < TILE_ELEMS; i += 256) {
    int tt = i % PADT; int r = i / PADT;
    int zz = r % PADZ; r /= PADZ;
    int yy = r % TLY;  int xx = r / TLY;
    tile[i] = ((const _Float16*)gpad)[lin0 + (size_t)(xx*PADY + yy)*YSTRIDE
                                            + zz*PADT + tt];
  }
  __syncthreads();
#endif

  int lane = threadIdx.x & 31;
  int wave = threadIdx.x >> 5;
  int hi   = lane >> 4;
  int n    = lane & 15;      // A row (voxel) / B,C column (channel)

  // ---- B fragments: W0 f32->f16; rows K>=81 and cols N>=8 zero ----
  // B 32x16 f16 layout: lane element e -> K = (e<8?e:e+8) + 8*hi, N = lane%16
  v16h bf[3];
  #pragma unroll
  for (int w = 0; w < 3; ++w) {
    v16h bb;
    #pragma unroll
    for (int e = 0; e < 16; ++e) {
      int kg = 32*w + ((e < 8) ? e : (e + 8)) + hi*8;
      float bw = 0.f;
      if (kg < 81 && n < 8) bw = W0[kg*CMID + n];
      bb[e] = (_Float16)bw;
    }
    bf[w] = bb;
  }

  // ---- lane-constant stencil offsets ----
  // A 16x32 f16 layout: lanes<16: e -> K = (e<8?e:e+8); lanes>=16: +8.
  // w=0,1: all K valid. w=2: e0..7 valid, e8 valid iff hi==0, e9..15 -> K>=81.
  int dof[3][16];
  #pragma unroll
  for (int w = 0; w < 3; ++w) {
    #pragma unroll
    for (int e = 0; e < 16; ++e) {
      int kl = (e < 8) ? e : (e + 8);
      int kg = 32*w + kl + hi*8;
      int d = 0;
      if (kg < 81) {
        int dt = kg % 3; int r2 = kg / 3;
        int dz = r2 % 3; r2 /= 3;
        int dy = r2 % 3; int dxk = r2 / 3;
        d = ((dxk*TLY + dy)*PADZ + dz)*PADT + dt;
      }
      dof[w][e] = d;
    }
  }

  // uniform head weights -> scalar loads into SGPRs
  float b0v[8], w1v[8];
  #pragma unroll
  for (int cc = 0; cc < 8; ++cc) { b0v[cc] = b0[cc]; w1v[cc] = W1[cc]; }
  float b1v = b1[0];

  float* scrw = scr + wave * (16 * SCR_STRIDE);
  size_t outBase = (size_t)b*VOX + (size_t)x*(DY*DZ*DT) + (size_t)y0*(DZ*DT);

  for (int it = 0; it < 10; ++it) {
    int tileBase = (wave*10 + it) * 16;

    // this lane's A-row voxel v = yl*80 + z*5 + t
    int vr = tileBase + n;
    int yl = vr / 80;
    int zt = vr % 80;
    int z  = zt / 5, t = zt % 5;
    int base = (yl*PADZ + z)*PADT + t;   // halo folded into dof[]

    v16h a0, a1, a2;
    #pragma unroll
    for (int e = 0; e < 16; ++e) a0[e] = tile[base + dof[0][e]];
    #pragma unroll
    for (int e = 0; e < 16; ++e) a1[e] = tile[base + dof[1][e]];
    #pragma unroll
    for (int e = 9; e < 16; ++e) a2[e] = (_Float16)0.f;   // K=81..95 pads
    #pragma unroll
    for (int e = 0; e < 8; ++e)  a2[e] = tile[base + dof[2][e]];
    { _Float16 t8 = tile[base + dof[2][8]];
      a2[8] = hi ? (_Float16)0.f : t8; }                  // K=80 vs K=88

    v8f c = {};
    c = __builtin_amdgcn_wmma_f32_16x16x32_f16(false, a0, false, bf[0], (short)0, c, false, false);
    c = __builtin_amdgcn_wmma_f32_16x16x32_f16(false, a1, false, bf[1], (short)0, c, false, false);
    c = __builtin_amdgcn_wmma_f32_16x16x32_f16(false, a2, false, bf[2], (short)0, c, false, false);

    // ---- head via LDS transpose (same-wave DS ops are in-order; no barrier)
    // C layout: lane l, vgpr v -> (M = v + 8*hi, N = l%16 = channel).
    #pragma unroll
    for (int v = 0; v < 8; ++v)
      scrw[(8*hi + v)*SCR_STRIDE + n] = c[v];

    if (lane < 16) {                     // lane n finishes voxel row n
      const float4 r0 = *(const float4*)&scrw[n*SCR_STRIDE];
      const float4 r1 = *(const float4*)&scrw[n*SCR_STRIDE + 4];
      float ctr = (float)tile[base + 2402];   // own center: idx(1,yl+1,z+1,t+1)
      float msk = (ctr > 0.f) ? 1.f : 0.f;
      float rc[8] = {r0.x, r0.y, r0.z, r0.w, r1.x, r1.y, r1.z, r1.w};
      float acc = 0.f;
      #pragma unroll
      for (int cc = 0; cc < 8; ++cc) {
        float hv = fmaxf((rc[cc] + b0v[cc]) * msk, 0.f);
        acc = fmaf(hv, w1v[cc], acc);
      }
      out[outBase + tileBase + n] = (acc + b1v) * msk;   // coalesced 16-lane
    }
  }
}

extern "C" void kernel_launch(void* const* d_in, const int* in_sizes, int n_in,
                              void* d_out, int out_size, void* d_ws, size_t ws_size,
                              hipStream_t stream) {
  const float* pts = (const float*)d_in[0];   // (B,N,4)
  const float* W0  = (const float*)d_in[1];   // (81,1,8)
  const float* b0  = (const float*)d_in[2];   // (8)
  const float* W1  = (const float*)d_in[3];   // (8,1)
  const float* b1  = (const float*)d_in[4];   // (1)
  float* out = (float*)d_out;
  unsigned short* gpad = (unsigned short*)d_ws;   // ~32 MB padded f16 grid

  me_zero_grid<<<(GRID_U128 + 255)/256, 256, 0, stream>>>((uint4*)d_ws);

  int ntot = in_sizes[0] / 4;
  me_scatter<<<(ntot + 255)/256, 256, 0, stream>>>(pts, gpad, ntot);

  // one workgroup (8 wave32) per (b, x, 16-wide y strip)
  me_conv_mlp<<<BX*DX*(DY/16), 256, 0, stream>>>(gpad, W0, b0, W1, b1, out);
}
